// CSSSGNNModel_57269093925294
// MI455X (gfx1250) — matrix operand order repaired and checked by reading, not compiled, not run on hardware
//
#include <hip/hip_runtime.h>
#include <cstdint>
#include <cstddef>

#define NN 20000
#define NE 40000
#define NG 1024
#define DNF 64
#define DEF 16
#define DGF 32

typedef __bf16 bf16_t;
typedef bf16_t bf16x16 __attribute__((ext_vector_type(16)));
typedef float  f32x8   __attribute__((ext_vector_type(8)));

__device__ __forceinline__ unsigned short f2bf(float f) {
  unsigned u = __float_as_uint(f);
  unsigned r = ((u >> 16) & 1u) + 0x7FFFu;
  return (unsigned short)((u + r) >> 16);
}

// ---- WMMA fragment-packed layouts (ushort element offsets) ----
// A (M x K, K%32==0): block (mt,kt) holds a 16x32 fragment, 32 lanes x 16 contiguous bf16.
//   lane = (m&15) + 16*((kk>>3)&1) ; e = (kk&7) + 8*(kk>>4)
__device__ __forceinline__ size_t apack_off(int m, int k, int K) {
  int mt = m >> 4, ml = m & 15;
  int kt = k >> 5, kk = k & 31;
  int lane = ml + (((kk >> 3) & 1) << 4);
  int e = (kk & 7) + ((kk >> 4) << 3);
  return ((size_t)(mt * (K >> 5) + kt) * 32 + lane) * 16 + e;
}
// B (K x N): block (kt,nt) holds a 32x16 fragment.
//   lane = (n&15) + 16*((kk>>3)&1) ; e = (kk&7) + 8*(kk>>4)
__device__ __forceinline__ size_t bpack_off(int k, int n, int N) {
  int kt = k >> 5, kk = k & 31;
  int nt = n >> 4, nl = n & 15;
  int lane = nl + (((kk >> 3) & 1) << 4);
  int e = (kk & 7) + ((kk >> 4) << 3);
  return ((size_t)(kt * (N >> 4) + nt) * 32 + lane) * 16 + e;
}

union FragU { bf16x16 v; uint4 q[2]; };

// ---------------- elementwise / reduction kernels ----------------

__global__ void fill0_k(float* p, int n) {
  int id = blockIdx.x * blockDim.x + threadIdx.x;
  if (id < n) p[id] = 0.f;
}

__global__ void count_k(const int* __restrict__ seg, int n, float* cnt) {
  int id = blockIdx.x * blockDim.x + threadIdx.x;
  if (id < n) atomicAdd(&cnt[seg[id]], 1.f);
}

__global__ void scatter_add_k(const float* __restrict__ in, const int* __restrict__ seg,
                              int M, int F, float* out) {
  int id = blockIdx.x * blockDim.x + threadIdx.x;
  if (id >= M * F) return;
  int i = id / F, f = id - i * F;
  atomicAdd(&out[(size_t)seg[i] * F + f], in[id]);
}

__global__ void div_cnt_k(float* x, const float* __restrict__ cnt, int total, int F) {
  int id = blockIdx.x * blockDim.x + threadIdx.x;
  if (id >= total) return;
  x[id] /= fmaxf(cnt[id / F], 1.f);
}

__global__ void bn_stats_k(const float* __restrict__ x, int rows, int F,
                           const float* __restrict__ g, const float* __restrict__ b,
                           float* scale, float* shift) {
  int f = blockIdx.x;
  float s = 0.f, s2 = 0.f;
  for (int r = threadIdx.x; r < rows; r += blockDim.x) {
    float v = x[(size_t)r * F + f];
    s += v; s2 += v * v;
  }
  __shared__ float sh[256], sh2[256];
  int t = threadIdx.x;
  sh[t] = s; sh2[t] = s2; __syncthreads();
  for (int st = 128; st > 0; st >>= 1) {
    if (t < st) { sh[t] += sh[t + st]; sh2[t] += sh2[t + st]; }
    __syncthreads();
  }
  if (t == 0) {
    float m = sh[0] / rows;
    float var = sh2[0] / rows - m * m;
    float sc = g[f] * rsqrtf(var + 1e-5f);
    scale[f] = sc;
    shift[f] = b[f] - m * sc;
  }
}

__global__ void bn_apply_k(const float* __restrict__ x, const float* __restrict__ scale,
                           const float* __restrict__ shift, float* y, int rows, int F) {
  int id = blockIdx.x * blockDim.x + threadIdx.x;
  if (id >= rows * F) return;
  int f = id % F;
  y[id] = x[id] * scale[f] + shift[f];
}

// up-to-3-segment gather/concat -> fragment-packed bf16 A matrix (M x Kpad)
__global__ void concat_gather_bf16_k(
    const float* __restrict__ s0, const int* __restrict__ i0, int w0,
    const float* __restrict__ s1, const int* __restrict__ i1, int w1,
    const float* __restrict__ s2, const int* __restrict__ i2, int w2,
    unsigned short* __restrict__ out, int M, int Kpad) {
  int id = blockIdx.x * blockDim.x + threadIdx.x;
  int total = M * Kpad;
  if (id >= total) return;
  int i = id / Kpad, k = id - i * Kpad;
  int k0 = k;
  float v = 0.f;
  if (k < w0) {
    int r = i0 ? i0[i] : i;
    v = s0[(size_t)r * w0 + k];
  } else {
    k -= w0;
    if (s1 && k < w1) {
      int r = i1 ? i1[i] : i;
      v = s1[(size_t)r * w1 + k];
    } else {
      k -= w1;
      if (s2 && k < w2) {
        int r = i2 ? i2[i] : i;
        v = s2[(size_t)r * w2 + k];
      }
    }
  }
  out[apack_off(i, k0, Kpad)] = f2bf(v);
}

// weight f32 (K x Nc) -> fragment-packed bf16 (Kpad x Nc), zero padded rows
__global__ void conv_w_k(const float* __restrict__ w, unsigned short* out,
                         int K, int Nc, int Kpad) {
  int id = blockIdx.x * blockDim.x + threadIdx.x;
  if (id >= Kpad * Nc) return;
  int k = id / Nc, n = id - k * Nc;
  out[bpack_off(k, n, Nc)] = (k < K) ? f2bf(w[(size_t)k * Nc + n]) : (unsigned short)0;
}

// final head: sigmoid(concat([rgo,pgo]) @ w + b) * 100
__global__ void ylin_k(const float* __restrict__ rgo, const float* __restrict__ pgo,
                       const float* __restrict__ w, const float* __restrict__ b,
                       float* out, int Gn) {
  int g = blockIdx.x * blockDim.x + threadIdx.x;
  if (g >= Gn) return;
  float acc = b[0];
  for (int j = 0; j < 256; ++j) acc += rgo[(size_t)g * 256 + j] * w[j];
  for (int j = 0; j < 256; ++j) acc += pgo[(size_t)g * 256 + j] * w[256 + j];
  out[g] = 100.f / (1.f + __expf(-acc));
}

// ---------------- WMMA bf16 GEMM (LDS-free, fragment-packed operands) ----------------
// C(MxN) = A(MxK) * B(KxN) + bias ; act: 0 none, 1 relu, 2 elu
// A in apack layout, B in bpack layout. 256 threads = 8 waves; each wave owns a
// 16(M) x 64(N) tile; per K-step: 2 b128 A loads + 8 b128 B loads + 4 WMMAs.
// Requires: M%16==0, N%64==0, K%32==0 (all call sites guarantee this).

__global__ __launch_bounds__(256) void gemm_bf16_k(
    const unsigned short* __restrict__ A, const unsigned short* __restrict__ B,
    const float* __restrict__ bias, float* __restrict__ Cf,
    unsigned short* __restrict__ Cb, int M, int N, int K, int act) {
  const int t = threadIdx.x;
  const int lane = t & 31, wave = t >> 5;
  const int Mtiles = M >> 4;
  const int mt = blockIdx.y * 8 + wave;
  if (mt >= Mtiles) return;              // wave-uniform exit, no barriers in kernel
  const int n0t = blockIdx.x * 4;        // base N subtile (16-wide units)
  const int Ktiles = K >> 5;
  const int Ntiles = N >> 4;

  f32x8 acc[4];
#pragma unroll
  for (int i = 0; i < 4; ++i)
#pragma unroll
    for (int e = 0; e < 8; ++e) acc[i][e] = 0.f;

  const uint4* Ab = (const uint4*)A;
  const uint4* Bb = (const uint4*)B;

  for (int kt = 0; kt < Ktiles; ++kt) {
    FragU a;
    {
      const uint4* pa = Ab + (((size_t)mt * Ktiles + kt) * 32 + lane) * 2;
      a.q[0] = pa[0];
      a.q[1] = pa[1];
    }
    if (kt + 1 < Ktiles)
      __builtin_prefetch(Ab + (((size_t)mt * Ktiles + kt + 1) * 32 + lane) * 2, 0, 3);
#pragma unroll
    for (int nt = 0; nt < 4; ++nt) {
      FragU b;
      const uint4* pb = Bb + (((size_t)kt * Ntiles + n0t + nt) * 32 + lane) * 2;
      b.q[0] = pb[0];
      b.q[1] = pb[1];
      acc[nt] = __builtin_amdgcn_wmma_f32_16x16x32_bf16(false, a.v, false, b.v, (short)0,
                                                        acc[nt], false, false);
    }
  }

  // epilogue: C layout — VGPR v: lanes 0-15 -> M=v, lanes 16-31 -> M=v+8; N = lane%16
  const int cn = lane & 15;
  const int rml = (lane < 16) ? 0 : 8;
#pragma unroll
  for (int nt = 0; nt < 4; ++nt) {
    int col = (n0t + nt) * 16 + cn;
    float bv = bias ? bias[col] : 0.f;
#pragma unroll
    for (int v = 0; v < 8; ++v) {
      int row = (mt << 4) + rml + v;
      float val = acc[nt][v] + bv;
      if (act == 1) val = fmaxf(val, 0.f);
      else if (act == 2) val = (val > 0.f) ? val : expm1f(val);
      if (Cf) Cf[(size_t)row * N + col] = val;
      if (Cb) Cb[apack_off(row, col, N)] = f2bf(val);  // packed for next GEMM's A
    }
  }
}

// ---------------- host orchestration ----------------

static inline int cdiv_h(int a, int b) { return (a + b - 1) / b; }
static inline int rup32(int x) { return (x + 31) & ~31; }

extern "C" void kernel_launch(void* const* d_in, const int* in_sizes, int n_in,
                              void* d_out, int out_size, void* d_ws, size_t ws_size,
                              hipStream_t stream) {
  (void)in_sizes; (void)n_in; (void)out_size; (void)ws_size;

  const float* rx = (const float*)d_in[0];
  const float* re = (const float*)d_in[1];
  const int*   rc = (const int*)d_in[2];
  const int*   rb = (const int*)d_in[3];
  const float* rg = (const float*)d_in[4];
  const float* px = (const float*)d_in[5];
  const float* pe = (const float*)d_in[6];
  const int*   pc = (const int*)d_in[7];
  const int*   pb = (const int*)d_in[8];
  const float* bn_ng = (const float*)d_in[9];
  const float* bn_nb = (const float*)d_in[10];
  const float* bn_eg = (const float*)d_in[11];
  const float* bn_eb = (const float*)d_in[12];

  // per-layer input widths
  const int einA[8] = {144, 768, 768, 768, 768, 144, 768, 768};
  const int n1A[8]  = {576, 640, 640, 640, 640, 576, 640, 640};
  const int n2A[8]  = {192, 256, 256, 256, 256, 192, 256, 256};
  const int giA[8]  = {160, 256, 256, 256, 256, 128, 256, 256};
  const int goA[8]  = {128, 128, 128, 128, 256, 128, 128, 256};

  char* wsp = (char*)d_ws;
  size_t off = 0;
  auto alloc = [&](size_t bytes) -> void* {
    void* p = wsp + off;
    off += (bytes + 255) & ~(size_t)255;
    return p;
  };

  // ---- convert weights to fragment-packed bf16 ----
  unsigned short* W1[8][4]; unsigned short* W2[8][4];
  const float* B1[8][4]; const float* B2[8][4];
  for (int L = 0; L < 8; ++L) {
    int ks[4] = {einA[L], n1A[L], n2A[L], giA[L]};
    int hs[4] = {512, 128, 128, goA[L]};
    for (int j = 0; j < 4; ++j) {
      int k = ks[j], h = hs[j], kp = rup32(k);
      int bi = 13 + 16 * L + 4 * j;
      const float* w1f = (const float*)d_in[bi + 0];
      B1[L][j] = (const float*)d_in[bi + 1];
      const float* w2f = (const float*)d_in[bi + 2];
      B2[L][j] = (const float*)d_in[bi + 3];
      W1[L][j] = (unsigned short*)alloc((size_t)kp * h * 2);
      W2[L][j] = (unsigned short*)alloc((size_t)h * h * 2);
      conv_w_k<<<cdiv_h(kp * h, 256), 256, 0, stream>>>(w1f, W1[L][j], k, h, kp);
      conv_w_k<<<cdiv_h(h * h, 256), 256, 0, stream>>>(w2f, W2[L][j], h, h, h);
    }
  }
  unsigned short* RLW = (unsigned short*)alloc(256 * 256 * 2);
  unsigned short* PLW = (unsigned short*)alloc(256 * 256 * 2);
  conv_w_k<<<cdiv_h(256 * 256, 256), 256, 0, stream>>>((const float*)d_in[141], RLW, 256, 256, 256);
  conv_w_k<<<cdiv_h(256 * 256, 256), 256, 0, stream>>>((const float*)d_in[143], PLW, 256, 256, 256);
  const float* RLB = (const float*)d_in[142];
  const float* PLB = (const float*)d_in[144];
  const float* YW  = (const float*)d_in[145];
  const float* YB  = (const float*)d_in[146];

  // ---- activation scratch (reused across layers and sides) ----
  unsigned short* Abuf = (unsigned short*)alloc((size_t)NE * 768 * 2);
  unsigned short* Hbuf = (unsigned short*)alloc((size_t)NE * 512 * 2);
  float* Ebuf = (float*)alloc((size_t)NE * 512 * 4);
  float* Mbuf = (float*)alloc((size_t)NE * 128 * 4);
  float* AGG  = (float*)alloc((size_t)NN * 128 * 4);
  float* Xb0  = (float*)alloc((size_t)NN * 128 * 4);
  float* Xb1  = (float*)alloc((size_t)NN * 128 * 4);
  float* XG   = (float*)alloc((size_t)NG * 128 * 4);
  float* Ub0  = (float*)alloc((size_t)NG * 256 * 4);
  float* Ub1  = (float*)alloc((size_t)NG * 256 * 4);
  float* rgo  = (float*)alloc((size_t)NG * 256 * 4);
  float* pgo  = (float*)alloc((size_t)NG * 256 * 4);
  float* scale = (float*)alloc(64 * 4);
  float* shift = (float*)alloc(64 * 4);
  float* cntE  = (float*)alloc((size_t)NN * 4);
  float* cntN  = (float*)alloc((size_t)NG * 4);

  auto gemm = [&](const unsigned short* A, const unsigned short* B, const float* bias,
                  float* Cf, unsigned short* Cb, int M, int Nc, int K, int act) {
    dim3 grid(Nc / 64, cdiv_h(M >> 4, 8));
    gemm_bf16_k<<<grid, 256, 0, stream>>>(A, B, bias, Cf, Cb, M, Nc, K, act);
  };
  auto concat3 = [&](const float* s0, const int* i0, int w0,
                     const float* s1, const int* i1, int w1,
                     const float* s2, const int* i2, int w2,
                     unsigned short* out, int M, int Kpad) {
    int total = M * Kpad;
    concat_gather_bf16_k<<<cdiv_h(total, 256), 256, 0, stream>>>(
        s0, i0, w0, s1, i1, w1, s2, i2, w2, out, M, Kpad);
  };

  // ---- one graph side: BN, Lcnt meta layers, output linear (elu) ----
  auto side = [&](const float* xin, const float* efeat, const int* conn, const int* batch,
                  const float* u0, int Du0, int L0, int Lcnt,
                  const unsigned short* linW, const float* linB, float* go_out) {
    const int* rowI = conn;
    const int* colI = conn + NE;

    bn_stats_k<<<DNF, 256, 0, stream>>>(xin, NN, DNF, bn_ng, bn_nb, scale, shift);
    bn_apply_k<<<cdiv_h(NN * DNF, 256), 256, 0, stream>>>(xin, scale, shift, Xb0, NN, DNF);
    bn_stats_k<<<DEF, 256, 0, stream>>>(efeat, NE, DEF, bn_eg, bn_eb, scale, shift);
    bn_apply_k<<<cdiv_h(NE * DEF, 256), 256, 0, stream>>>(efeat, scale, shift, Ebuf, NE, DEF);

    fill0_k<<<cdiv_h(NN, 256), 256, 0, stream>>>(cntE, NN);
    count_k<<<cdiv_h(NE, 256), 256, 0, stream>>>(colI, NE, cntE);
    fill0_k<<<cdiv_h(NG, 256), 256, 0, stream>>>(cntN, NG);
    count_k<<<cdiv_h(NN, 256), 256, 0, stream>>>(batch, NN, cntN);

    float* xcur = Xb0; float* xalt = Xb1;
    const float* ucur = u0; int Du = Du0;
    float* ub[2] = {Ub0, Ub1}; int uidx = 0;
    int Dx = DNF, De = DEF;

    for (int li = 0; li < Lcnt; ++li) {
      int L = L0 + li;
      // edge model
      int einP = rup32(einA[L]);
      concat3(xcur, rowI, Dx, xcur, colI, Dx, Ebuf, nullptr, De, Abuf, NE, einP);
      gemm(Abuf, W1[L][0], B1[L][0], nullptr, Hbuf, NE, 512, einP, 1);
      gemm(Hbuf, W2[L][0], B2[L][0], Ebuf, nullptr, NE, 512, 512, 0);
      De = 512;
      // node message
      int n1P = rup32(n1A[L]);
      concat3(xcur, rowI, Dx, Ebuf, nullptr, 512, nullptr, nullptr, 0, Abuf, NE, n1P);
      gemm(Abuf, W1[L][1], B1[L][1], nullptr, Hbuf, NE, 128, n1P, 1);
      gemm(Hbuf, W2[L][1], B2[L][1], Mbuf, nullptr, NE, 128, 128, 0);
      // scatter-mean to nodes
      fill0_k<<<cdiv_h(NN * 128, 256), 256, 0, stream>>>(AGG, NN * 128);
      scatter_add_k<<<cdiv_h(NE * 128, 256), 256, 0, stream>>>(Mbuf, colI, NE, 128, AGG);
      div_cnt_k<<<cdiv_h(NN * 128, 256), 256, 0, stream>>>(AGG, cntE, NN * 128, 128);
      // node update
      int n2P = rup32(n2A[L]);
      concat3(xcur, nullptr, Dx, AGG, nullptr, 128, nullptr, nullptr, 0, Abuf, NN, n2P);
      gemm(Abuf, W1[L][2], B1[L][2], nullptr, Hbuf, NN, 128, n2P, 1);
      gemm(Hbuf, W2[L][2], B2[L][2], xalt, nullptr, NN, 128, 128, 0);
      { float* tsw = xcur; xcur = xalt; xalt = tsw; }
      Dx = 128;
      // graph mean + global model
      fill0_k<<<cdiv_h(NG * 128, 256), 256, 0, stream>>>(XG, NG * 128);
      scatter_add_k<<<cdiv_h(NN * 128, 256), 256, 0, stream>>>(xcur, batch, NN, 128, XG);
      div_cnt_k<<<cdiv_h(NG * 128, 256), 256, 0, stream>>>(XG, cntN, NG * 128, 128);
      int gP = rup32(giA[L]);
      if (ucur)
        concat3(ucur, nullptr, Du, XG, nullptr, 128, nullptr, nullptr, 0, Abuf, NG, gP);
      else
        concat3(XG, nullptr, 128, nullptr, nullptr, 0, nullptr, nullptr, 0, Abuf, NG, gP);
      int go = goA[L];
      gemm(Abuf, W1[L][3], B1[L][3], nullptr, Hbuf, NG, go, gP, 1);
      float* uo = ub[uidx]; uidx ^= 1;
      gemm(Hbuf, W2[L][3], B2[L][3], uo, nullptr, NG, go, go, 0);
      ucur = uo; Du = go;
    }
    // output linear with ELU
    concat3(ucur, nullptr, 256, nullptr, nullptr, 0, nullptr, nullptr, 0, Abuf, NG, 256);
    gemm(Abuf, linW, linB, go_out, nullptr, NG, 256, 256, 2);
  };

  // r side: 5 layers, u0 = rg
  side(rx, re, rc, rb, rg, DGF, 0, 5, RLW, RLB, rgo);
  // p side: 3 layers, no initial globals
  side(px, pe, pc, pb, nullptr, 0, 5, 3, PLW, PLB, pgo);

  ylin_k<<<cdiv_h(NG, 256), 256, 0, stream>>>(rgo, pgo, YW, YB, (float*)d_out, NG);
}
